// InterFusion_76115410420028
// MI455X (gfx1250) — matrix operand order repaired and checked by reading, not compiled, not run on hardware
//
#include <hip/hip_runtime.h>

// ---------------------------------------------------------------------------
// Problem constants (from the reference)
// ---------------------------------------------------------------------------
#define EMB   1024
#define NHID  64
#define HEADS 16
#define BATCH 8192
#define SEQ   2
#define ROWLEN (SEQ * EMB)          // 2048, layernorm is over [S,E] jointly
#define MROWS (BATCH * SEQ)         // 16384 rows for the attention GEMMs

typedef __bf16 bf16_t;
typedef float  v8f   __attribute__((ext_vector_type(8)));
typedef __bf16 v8bf  __attribute__((ext_vector_type(8)));
typedef __bf16 v16bf __attribute__((ext_vector_type(16)));

union BFrag { v16bf v; v8bf h[2]; };

// ---------------------------------------------------------------------------
// fp32 -> bf16 convert (weights)
// ---------------------------------------------------------------------------
__global__ void f32_to_bf16_kernel(const float* __restrict__ s,
                                   bf16_t* __restrict__ d, int n) {
    int i = blockIdx.x * blockDim.x + threadIdx.x;
    if (i < n) d[i] = (bf16_t)s[i];
}

// ---------------------------------------------------------------------------
// LayerNorm over the joint [S,E] = 2048 elements of each batch row.
// ---------------------------------------------------------------------------
__global__ __launch_bounds__(256)
void ln2d_kernel(const float* __restrict__ x, const float* __restrict__ g,
                 const float* __restrict__ bta, bf16_t* __restrict__ y) {
    __shared__ float ssum[256];
    __shared__ float ssq[256];
    const int b = blockIdx.x;
    const int t = threadIdx.x;
    const float* xr = x + (size_t)b * ROWLEN;
    float s = 0.f, q = 0.f;
#pragma unroll
    for (int i = 0; i < 8; ++i) {
        float v = xr[t + i * 256];
        s += v; q += v * v;
    }
    ssum[t] = s; ssq[t] = q;
    __syncthreads();
    for (int off = 128; off > 0; off >>= 1) {
        if (t < off) { ssum[t] += ssum[t + off]; ssq[t] += ssq[t + off]; }
        __syncthreads();
    }
    const float m   = ssum[0] * (1.f / ROWLEN);
    const float var = ssq[0] * (1.f / ROWLEN) - m * m;
    const float rs  = rsqrtf(var + 1e-5f);
    bf16_t* yr = y + (size_t)b * ROWLEN;
#pragma unroll
    for (int i = 0; i < 8; ++i) {
        int j = t + i * 256;
        yr[j] = (bf16_t)((xr[j] - m) * rs * g[j] + bta[j]);
    }
}

// ---------------------------------------------------------------------------
// WMMA bf16 GEMM:  C[m,n] = act( A[m,:] . W[n,:] + bias[n] ) (+ Res)
//   A : (M,K) bf16 row-major, row stride lda;  W : (N,K) bf16 row-major
// Block tile 128(M) x 128(N), 8 waves (4Mx2N), each wave 32x64 = 2x4 WMMA
// tiles of v_wmma_f32_16x16x32_bf16.  A tile is double-buffered in LDS via
// GLOBAL_LOAD_ASYNC_TO_LDS_B128 (ASYNCcnt), B fragments stream from global
// (weights are L2-resident: 192MB L2 >> 24MB of bf16 weights).
// ---------------------------------------------------------------------------
__global__ __launch_bounds__(256)
void gemm_bf16_wmma(const bf16_t* __restrict__ A, int lda,
                    const bf16_t* __restrict__ W,
                    const float* __restrict__ bias,
                    float* __restrict__ Cf, bf16_t* __restrict__ Cb, int ldc,
                    const float* __restrict__ Res, int ldr,
                    int M, int N, int K, int act_tanh) {
    __shared__ bf16_t As[2][128 * 40];       // double-buffered 128x32, pad->40
    const int tid  = threadIdx.x;
    const int lane = tid & 31;
    const int wave = tid >> 5;
    const int wm = wave >> 1;                // 0..3 : wave M position (x32)
    const int wn = wave & 1;                 // 0..1 : wave N position (x64)
    const int tileN = blockIdx.x * 128;
    const int tileM = blockIdx.y * 128;

    v8f acc[8] = {};                         // 2(M) x 4(N) tiles of 16x16 f32

    const int half = lane >> 4;              // 0/1: lane half
    const int lid  = lane & 15;
    const int am0  = wm * 32 + lid;          // A-frag row (a0); a1 = +16
    const int kb8  = half * 8;               // A-frag K sub-base (ISA layout)

    // B fragments: 2x8 bf16 per lane from W row n (k 8h..8h+7, 16+8h..16+8h+7)
    const bf16_t* wrow[4];
#pragma unroll
    for (int j = 0; j < 4; ++j)
        wrow[j] = W + (size_t)(tileN + wn * 64 + j * 16 + lid) * K + half * 8;

    // Per-thread A chunks: 512 16B chunks per 128x32 tile, 2 per thread
    const int c_col = (tid & 3) * 8;
    const bf16_t* aptr[2];
    unsigned ldsOff[2][2];                   // [buffer][chunk]
#pragma unroll
    for (int c = 0; c < 2; ++c) {
        const int row = (tid + c * 256) >> 2;
        aptr[c] = A + (size_t)(tileM + row) * lda + c_col;
#pragma unroll
        for (int bi = 0; bi < 2; ++bi)
            ldsOff[bi][c] = (unsigned)(uintptr_t)&As[bi][row * 40 + c_col];
    }

    // Prologue: async-copy tile k=0 into buffer 0 (memory -> LDS, no VGPRs)
#pragma unroll
    for (int c = 0; c < 2; ++c)
        asm volatile("global_load_async_to_lds_b128 %0, %1, off"
                     :: "v"(ldsOff[0][c]),
                        "v"((unsigned long long)(uintptr_t)aptr[c])
                     : "memory");
    asm volatile("s_wait_asynccnt 0x0" ::: "memory");
    __syncthreads();

    for (int k0 = 0; k0 < K; k0 += 32) {
        const int cur = (k0 >> 5) & 1;
        const int nxt = cur ^ 1;

        // Kick off async copy of the NEXT A tile while we compute this one.
        if (k0 + 32 < K) {
#pragma unroll
            for (int c = 0; c < 2; ++c)
                asm volatile("global_load_async_to_lds_b128 %0, %1, off"
                             :: "v"(ldsOff[nxt][c]),
                                "v"((unsigned long long)(uintptr_t)(aptr[c] + k0 + 32))
                             : "memory");
        }

        // Near-scope prefetch of upcoming B stream (global_prefetch_b8)
        __builtin_prefetch(wrow[0] + k0 + 32, 0, 3);
        __builtin_prefetch(wrow[2] + k0 + 32, 0, 3);

        const bf16_t* base = &As[cur][0];
        BFrag a0, a1, b[4];
        a0.h[0] = *(const v8bf*)&base[(am0)      * 40 + kb8];
        a0.h[1] = *(const v8bf*)&base[(am0)      * 40 + kb8 + 16];
        a1.h[0] = *(const v8bf*)&base[(am0 + 16) * 40 + kb8];
        a1.h[1] = *(const v8bf*)&base[(am0 + 16) * 40 + kb8 + 16];
#pragma unroll
        for (int j = 0; j < 4; ++j) {
            const bf16_t* wp = wrow[j] + k0;
            b[j].h[0] = *(const v8bf*)(wp);
            b[j].h[1] = *(const v8bf*)(wp + 16);
        }

#pragma unroll
        for (int j = 0; j < 4; ++j) {
            acc[j] = __builtin_amdgcn_wmma_f32_16x16x32_bf16(
                false, a0.v, false, b[j].v, (short)0, acc[j], false, false);
            acc[4 + j] = __builtin_amdgcn_wmma_f32_16x16x32_bf16(
                false, a1.v, false, b[j].v, (short)0, acc[4 + j], false, false);
        }

        // Drain our async copy, then block-wide barrier before buffer swap.
        asm volatile("s_wait_asynccnt 0x0" ::: "memory");
        __syncthreads();
    }

    // Epilogue: C/D 16x16 layout -> lane half selects M+8, VGPR r is M row
#pragma unroll
    for (int mi = 0; mi < 2; ++mi) {
#pragma unroll
        for (int j = 0; j < 4; ++j) {
            const int n  = tileN + wn * 64 + j * 16 + lid;
            const float bv = bias ? bias[n] : 0.f;
            const v8f a = acc[mi * 4 + j];
#pragma unroll
            for (int r = 0; r < 8; ++r) {
                const int m = tileM + wm * 32 + mi * 16 + r + half * 8;
                float val = a[r] + bv;
                if (act_tanh) val = tanhf(val);
                if (Res) val += Res[(size_t)m * ldr + n];
                if (Cf)  Cf[(size_t)m * ldc + n] = val;
                else     Cb[(size_t)m * ldc + n] = (bf16_t)val;
            }
        }
    }
}

// ---------------------------------------------------------------------------
// S=2 attention; head index is the FAST dim: q[b,h,s,d] = Qproj[b,s,d*16+h].
// ---------------------------------------------------------------------------
__global__ __launch_bounds__(256)
void attn_s2_kernel(const bf16_t* __restrict__ Q, const bf16_t* __restrict__ Km,
                    const bf16_t* __restrict__ V, bf16_t* __restrict__ O) {
    const int gid = blockIdx.x * blockDim.x + threadIdx.x;   // [0, B*H)
    const int b = gid >> 4;
    const int h = gid & 15;
    const size_t base = (size_t)b * SEQ * EMB + h;
    const bf16_t* q0 = Q + base;  const bf16_t* q1 = q0 + EMB;
    const bf16_t* k0 = Km + base; const bf16_t* k1 = k0 + EMB;
    float s00 = 0.f, s01 = 0.f, s10 = 0.f, s11 = 0.f;
#pragma unroll 8
    for (int d = 0; d < NHID; ++d) {
        const int i = d * HEADS;
        float a0 = (float)q0[i], a1 = (float)q1[i];
        float c0 = (float)k0[i], c1 = (float)k1[i];
        s00 += a0 * c0; s01 += a0 * c1;
        s10 += a1 * c0; s11 += a1 * c1;
    }
    const float sc = 0.125f;                 // 1/sqrt(64)
    s00 *= sc; s01 *= sc; s10 *= sc; s11 *= sc;
    const float m0 = fmaxf(s00, s01), m1 = fmaxf(s10, s11);
    const float e00 = __expf(s00 - m0), e01 = __expf(s01 - m0);
    const float e10 = __expf(s10 - m1), e11 = __expf(s11 - m1);
    const float r0 = 1.f / (e00 + e01), r1 = 1.f / (e10 + e11);
    const float p00 = e00 * r0, p01 = e01 * r0;
    const float p10 = e10 * r1, p11 = e11 * r1;
    const bf16_t* v0 = V + base; const bf16_t* v1 = v0 + EMB;
    bf16_t* o0 = O + base; bf16_t* o1 = o0 + EMB;
#pragma unroll 8
    for (int d = 0; d < NHID; ++d) {
        const int i = d * HEADS;
        float w0 = (float)v0[i], w1 = (float)v1[i];
        o0[i] = (bf16_t)(p00 * w0 + p01 * w1);
        o1[i] = (bf16_t)(p10 * w0 + p11 * w1);
    }
}

// ---------------------------------------------------------------------------
// Host-side orchestration (graph-capture safe: only kernel launches on stream)
// ---------------------------------------------------------------------------
extern "C" void kernel_launch(void* const* d_in, const int* in_sizes, int n_in,
                              void* d_out, int out_size, void* d_ws, size_t ws_size,
                              hipStream_t stream) {
    (void)in_sizes; (void)n_in; (void)out_size; (void)ws_size;
    const float* input = (const float*)d_in[0];
    const float* Wq    = (const float*)d_in[1];
    const float* Wk    = (const float*)d_in[2];
    const float* Wv    = (const float*)d_in[3];
    const float* Wo    = (const float*)d_in[4];
    const float* ln1w  = (const float*)d_in[5];
    const float* ln1b  = (const float*)d_in[6];
    const float* ln2w  = (const float*)d_in[7];
    const float* ln2b  = (const float*)d_in[8];
    const float* f1w1  = (const float*)d_in[9];
    const float* f1b1  = (const float*)d_in[10];
    const float* f1w2  = (const float*)d_in[11];
    const float* f1b2  = (const float*)d_in[12];
    const float* f2w1  = (const float*)d_in[13];
    const float* f2b1  = (const float*)d_in[14];
    const float* f2w2  = (const float*)d_in[15];
    const float* f2b2  = (const float*)d_in[16];
    float* out = (float*)d_out;

    const size_t MB = 1ull << 20;
    char* wsb = (char*)d_ws;
    bf16_t* wq_b   = (bf16_t*)(wsb + 0 * MB);
    bf16_t* wk_b   = (bf16_t*)(wsb + 2 * MB);
    bf16_t* wv_b   = (bf16_t*)(wsb + 4 * MB);
    bf16_t* wo_b   = (bf16_t*)(wsb + 6 * MB);
    bf16_t* f1w1_b = (bf16_t*)(wsb + 8 * MB);
    bf16_t* f1w2_b = (bf16_t*)(wsb + 12 * MB);
    bf16_t* f2w1_b = (bf16_t*)(wsb + 16 * MB);
    bf16_t* f2w2_b = (bf16_t*)(wsb + 20 * MB);
    bf16_t* xs     = (bf16_t*)(wsb + 24 * MB);   // (16384,1024); reused as concat
    bf16_t* qb     = (bf16_t*)(wsb + 56 * MB);   // (16384,1024); reused as out_s
    bf16_t* kb     = (bf16_t*)(wsb + 88 * MB);   // (16384,1024); reused as ffn hidden
    bf16_t* vb     = (bf16_t*)(wsb + 120 * MB);  // (16384,1024)
    float*  out1   = (float*)(wsb + 152 * MB);   // (16384,1024) f32, 64 MB

    const dim3 blk(256);

    // 1) weight downconvert fp32 -> bf16
    f32_to_bf16_kernel<<<(EMB * EMB) / 256, blk, 0, stream>>>(Wq, wq_b, EMB * EMB);
    f32_to_bf16_kernel<<<(EMB * EMB) / 256, blk, 0, stream>>>(Wk, wk_b, EMB * EMB);
    f32_to_bf16_kernel<<<(EMB * EMB) / 256, blk, 0, stream>>>(Wv, wv_b, EMB * EMB);
    f32_to_bf16_kernel<<<(EMB * EMB) / 256, blk, 0, stream>>>(Wo, wo_b, EMB * EMB);
    f32_to_bf16_kernel<<<(2 * EMB * EMB) / 256, blk, 0, stream>>>(f1w1, f1w1_b, 2 * EMB * EMB);
    f32_to_bf16_kernel<<<(2 * EMB * EMB) / 256, blk, 0, stream>>>(f1w2, f1w2_b, 2 * EMB * EMB);
    f32_to_bf16_kernel<<<(2 * EMB * EMB) / 256, blk, 0, stream>>>(f2w1, f2w1_b, 2 * EMB * EMB);
    f32_to_bf16_kernel<<<(2 * EMB * EMB) / 256, blk, 0, stream>>>(f2w2, f2w2_b, 2 * EMB * EMB);

    // 2) LN1 over [S,E] -> xs (bf16)
    ln2d_kernel<<<BATCH, blk, 0, stream>>>(input, ln1w, ln1b, xs);

    // 3) Q,K,V projections: (16384,1024) x (1024,1024)^T
    const dim3 gproj(EMB / 128, MROWS / 128);
    gemm_bf16_wmma<<<gproj, blk, 0, stream>>>(xs, EMB, wq_b, nullptr,
                                              nullptr, qb, EMB, nullptr, 0,
                                              MROWS, EMB, EMB, 0);
    gemm_bf16_wmma<<<gproj, blk, 0, stream>>>(xs, EMB, wk_b, nullptr,
                                              nullptr, kb, EMB, nullptr, 0,
                                              MROWS, EMB, EMB, 0);
    gemm_bf16_wmma<<<gproj, blk, 0, stream>>>(xs, EMB, wv_b, nullptr,
                                              nullptr, vb, EMB, nullptr, 0,
                                              MROWS, EMB, EMB, 0);

    // 4) attention (S=2, per (b,h)); writes concat layout into xs (reuse)
    attn_s2_kernel<<<(BATCH * HEADS) / 256, blk, 0, stream>>>(qb, kb, vb, xs);

    // 5) O-projection + residual(input) -> out1 (f32)
    gemm_bf16_wmma<<<gproj, blk, 0, stream>>>(xs, EMB, wo_b, nullptr,
                                              out1, nullptr, EMB, input, EMB,
                                              MROWS, EMB, EMB, 0);

    // 6) LN2 -> out_s (bf16) into qb (reuse)
    ln2d_kernel<<<BATCH, blk, 0, stream>>>(out1, ln2w, ln2b, qb);

    // 7) FFN per position: rows of pos p are at offset p*E with stride 2*E
    const dim3 gffn1((2 * EMB) / 128, BATCH / 128);  // (8192,2048) out
    const dim3 gffn2(EMB / 128, BATCH / 128);        // (8192,1024) out
    // position 0 (f1)
    gemm_bf16_wmma<<<gffn1, blk, 0, stream>>>(qb, ROWLEN, f1w1_b, f1b1,
                                              nullptr, kb, 2 * EMB, nullptr, 0,
                                              BATCH, 2 * EMB, EMB, 1);
    gemm_bf16_wmma<<<gffn2, blk, 0, stream>>>(kb, 2 * EMB, f1w2_b, f1b2,
                                              out, nullptr, ROWLEN, out1, ROWLEN,
                                              BATCH, EMB, 2 * EMB, 1);
    // position 1 (f2)
    gemm_bf16_wmma<<<gffn1, blk, 0, stream>>>(qb + EMB, ROWLEN, f2w1_b, f2b1,
                                              nullptr, kb, 2 * EMB, nullptr, 0,
                                              BATCH, 2 * EMB, EMB, 1);
    gemm_bf16_wmma<<<gffn2, blk, 0, stream>>>(kb, 2 * EMB, f2w2_b, f2b2,
                                              out + EMB, nullptr, ROWLEN, out1 + EMB, ROWLEN,
                                              BATCH, EMB, 2 * EMB, 1);
}
